// GATBlock_73023033967279
// MI455X (gfx1250) — matrix operand order repaired and checked by reading, not compiled, not run on hardware
//
#include <hip/hip_runtime.h>
#include <hip/hip_bf16.h>

#define NN 100000
#define EE 800000
#define DDIM 64
#define HEADS 4

typedef __attribute__((ext_vector_type(16))) __bf16 v16bf;
typedef __attribute__((ext_vector_type(8)))  __bf16 v8bf;
typedef __attribute__((ext_vector_type(8)))  float  v8f;

__device__ __forceinline__ v16bf concat16(v8bf lo, v8bf hi) {
  return __builtin_shufflevector(lo, hi, 0, 1, 2, 3, 4, 5, 6, 7,
                                 8, 9, 10, 11, 12, 13, 14, 15);
}

// Monotone float <-> uint encoding for atomicMax-based segment max.
__device__ __forceinline__ unsigned fenc(float f) {
  unsigned u = __float_as_uint(f);
  return (u & 0x80000000u) ? ~u : (u | 0x80000000u);
}
__device__ __forceinline__ float fdec(unsigned e) {
  return (e & 0x80000000u) ? __uint_as_float(e ^ 0x80000000u)
                           : __uint_as_float(~e);
}
#define ENC_NEG_INF 0x007FFFFFu   // fenc(-inf)

// ---------------------------------------------------------------------------
// Kernel 0: out = bias (broadcast), amax = -inf, denom = 0
// ---------------------------------------------------------------------------
__global__ __launch_bounds__(256)
void init_kernel(float* __restrict__ out, const float* __restrict__ bias,
                 unsigned* __restrict__ amax, float* __restrict__ denom) {
  int i = blockIdx.x * 256 + threadIdx.x;
  if (i < NN * DDIM) out[i] = bias[i & 63];
  if (i < NN * HEADS) { amax[i] = ENC_NEG_INF; denom[i] = 0.0f; }
}

// ---------------------------------------------------------------------------
// Kernel 1: fused LayerNorm + GEMM  h = LN(x) @ W + b   (N x 64) @ (64 x 256)
// One block = 16-node M tile. 8 waves, each wave computes two 16x16 N tiles
// via v_wmma_f32_16x16x32_bf16 (two K=32 steps cover K=64).
// LDS layouts are fragment-contiguous so loads are ds_load_b128, not u16.
// ---------------------------------------------------------------------------
__global__ __launch_bounds__(256)
void ln_wmma_gemm(const float* __restrict__ x, const float* __restrict__ gamma,
                  const float* __restrict__ beta, const float* __restrict__ W,
                  const float* __restrict__ bvec, float* __restrict__ h) {
  // A stored per-lane, fragment order: lane L holds its 32 bf16 elements
  // contiguously (eidx 0..15 = K-block 0..31, 16..31 = K-block 32..63).
  __shared__ __bf16 lA[32][40];     // 40-elem (80 B) rows: 16B-aligned, pad
  // W transposed: lW[n][k], k contiguous so B fragments are contiguous runs.
  __shared__ __bf16 lW[256][72];    // 72-elem (144 B) rows: 16B-aligned, pad

  const int t = threadIdx.x;
  const int nodeBase = blockIdx.x * 16;

  // Stage W (row-major K x N in global, coalesced reads) transposed to LDS.
  for (int i = t; i < 64 * 256; i += 256) {
    const int k = i >> 8;     // 0..63
    const int n = i & 255;    // 0..255
    lW[n][k] = (__bf16)W[i];
  }

  // LayerNorm: 16 lanes per row, 4 elements per lane; shfl-xor reduce within
  // 16-lane groups (a row's lanes are contiguous inside a wave32). Results
  // are scattered directly into the WMMA A-fragment layout:
  //   lane = 16*((K>>4)&1) + row ; eidx = (K>>5)*16 + (K&15)
  {
    const int r = t >> 4;
    const int c = (t & 15) * 4;
    const float* xr = x + (size_t)(nodeBase + r) * 64 + c;
    float v0 = xr[0], v1 = xr[1], v2 = xr[2], v3 = xr[3];
    float s  = v0 + v1 + v2 + v3;
    float s2 = v0 * v0 + v1 * v1 + v2 * v2 + v3 * v3;
    for (int o = 8; o; o >>= 1) {
      s  += __shfl_xor(s,  o, 16);
      s2 += __shfl_xor(s2, o, 16);
    }
    const float mu  = s * (1.0f / 64.0f);
    const float var = s2 * (1.0f / 64.0f) - mu * mu;
    const float rs  = rsqrtf(var + 1e-5f);
    float y[4] = {(v0 - mu) * rs * gamma[c + 0] + beta[c + 0],
                  (v1 - mu) * rs * gamma[c + 1] + beta[c + 1],
                  (v2 - mu) * rs * gamma[c + 2] + beta[c + 2],
                  (v3 - mu) * rs * gamma[c + 3] + beta[c + 3]};
#pragma unroll
    for (int j = 0; j < 4; ++j) {
      const int K    = c + j;
      const int lane = (((K >> 4) & 1) << 4) + r;
      const int eidx = ((K >> 5) << 4) + (K & 15);
      lA[lane][eidx] = (__bf16)y[j];
    }
  }
  __syncthreads();

  const int wave = t >> 5;
  const int lane = t & 31;
  const int half = lane >> 4;
  const int m    = lane & 15;

  // A fragments: contiguous 2 x 16 bf16 per lane -> ds_load_b128 pairs.
  const v16bf a0 = concat16(*(const v8bf*)&lA[lane][0],
                            *(const v8bf*)&lA[lane][8]);
  const v16bf a1 = concat16(*(const v8bf*)&lA[lane][16],
                            *(const v8bf*)&lA[lane][24]);

#pragma unroll
  for (int tt = 0; tt < 2; ++tt) {
    const int tileN = wave + tt * 8;
    const int ncol  = tileN * 16 + m;
    // B fragment for lane: W[kbase + e][ncol], e=0..15, kbase=kstep*32+half*16
    // -> contiguous run in transposed LDS.
    const __bf16* col = &lW[ncol][half << 4];
    const v16bf b0 = concat16(*(const v8bf*)(col + 0),
                              *(const v8bf*)(col + 8));
    const v16bf b1 = concat16(*(const v8bf*)(col + 32),
                              *(const v8bf*)(col + 40));
    v8f c = {0.f, 0.f, 0.f, 0.f, 0.f, 0.f, 0.f, 0.f};
    c = __builtin_amdgcn_wmma_f32_16x16x32_bf16(false, a0, false, b0,
                                                (short)0, c, false, false);
    c = __builtin_amdgcn_wmma_f32_16x16x32_bf16(false, a1, false, b1,
                                                (short)0, c, false, false);
    // C layout: VGPR i -> row i + 8*half, col = lane&15 within tile.
    const float bc = bvec[ncol];
#pragma unroll
    for (int i = 0; i < 8; ++i) {
      const int row = i + half * 8;
      h[(size_t)(nodeBase + row) * 256 + ncol] = c[i] + bc;
    }
  }
}

// ---------------------------------------------------------------------------
// Kernel 2: per-edge GATv2 score + segment max (one wave per edge).
// alpha[e,h] = sum_d leakyrelu(hl[src]+hr[dst]+he)[h,d] * att[h,d]
// ---------------------------------------------------------------------------
__global__ __launch_bounds__(256)
void edge_score_kernel(const float* __restrict__ hl, const float* __restrict__ hr,
                       const int* __restrict__ ei, const float* __restrict__ ea,
                       const float* __restrict__ We, const float* __restrict__ att,
                       float* __restrict__ alpha, unsigned* __restrict__ amax) {
  __shared__ float sWe[512];   // W_e: 2 x 256
  __shared__ float sAtt[256];  // att: 4 x 64
  const int t = threadIdx.x;
  sAtt[t] = att[t];
  sWe[t] = We[t];
  sWe[256 + t] = We[256 + t];
  __syncthreads();

  const int e = blockIdx.x * 8 + (t >> 5);
  if (e >= EE) return;
  const int lane = t & 31;
  const int src = ei[e];
  const int dst = ei[EE + e];
  const float ea0 = ea[2 * e], ea1 = ea[2 * e + 1];
  const float* rowl = hl + (size_t)src * 256;
  const float* rowr = hr + (size_t)dst * 256;
  __builtin_prefetch(rowl, 0, 0);   // global_prefetch_b8
  __builtin_prefetch(rowr, 0, 0);

  // 8 contiguous channels per lane -> all within one head (64 ch / head).
  const int base = lane * 8;
  float part = 0.0f;
#pragma unroll
  for (int j = 0; j < 8; ++j) {
    const int idx = base + j;
    float mv = rowl[idx] + rowr[idx] + ea0 * sWe[idx] + ea1 * sWe[256 + idx];
    mv = mv > 0.0f ? mv : 0.2f * mv;
    part += mv * sAtt[idx];
  }
  part += __shfl_xor(part, 4, 8);
  part += __shfl_xor(part, 2, 8);
  part += __shfl_xor(part, 1, 8);
  if ((lane & 7) == 0) {
    const int hidx = lane >> 3;
    alpha[(size_t)e * 4 + hidx] = part;
    atomicMax(&amax[(size_t)dst * 4 + hidx], fenc(part));
  }
}

// ---------------------------------------------------------------------------
// Kernel 3: ex = exp(alpha - amax[dst]); denom[dst] += ex (one thread / (e,h))
// ---------------------------------------------------------------------------
__global__ __launch_bounds__(256)
void edge_exp_kernel(const int* __restrict__ ei, float* __restrict__ alpha,
                     const unsigned* __restrict__ amax, float* __restrict__ denom) {
  const int i = blockIdx.x * 256 + threadIdx.x;
  if (i >= EE * HEADS) return;
  const int e = i >> 2, hq = i & 3;
  const int dst = ei[EE + e];
  const float mx = fdec(amax[(size_t)dst * 4 + hq]);
  const float ex = __expf(alpha[i] - mx);
  alpha[i] = ex;
  atomicAdd(&denom[(size_t)dst * 4 + hq], ex);
}

// ---------------------------------------------------------------------------
// Kernel 4: aggregation with head-mean folded in (one wave per edge):
// out[dst,d] += sum_h hl[src,h,d] * (ex/(denom+eps)) / H
// ---------------------------------------------------------------------------
__global__ __launch_bounds__(256)
void edge_agg_kernel(const float* __restrict__ hl, const int* __restrict__ ei,
                     const float* __restrict__ alpha, const float* __restrict__ denom,
                     float* __restrict__ out) {
  const int t = threadIdx.x;
  const int e = blockIdx.x * 8 + (t >> 5);
  if (e >= EE) return;
  const int lane = t & 31;
  const int src = ei[e];
  const int dst = ei[EE + e];
  const float* row = hl + (size_t)src * 256;
  __builtin_prefetch(row, 0, 0);
  const float c0 = alpha[(size_t)e * 4 + 0] / (denom[(size_t)dst * 4 + 0] + 1e-16f) * 0.25f;
  const float c1 = alpha[(size_t)e * 4 + 1] / (denom[(size_t)dst * 4 + 1] + 1e-16f) * 0.25f;
  const float c2 = alpha[(size_t)e * 4 + 2] / (denom[(size_t)dst * 4 + 2] + 1e-16f) * 0.25f;
  const float c3 = alpha[(size_t)e * 4 + 3] / (denom[(size_t)dst * 4 + 3] + 1e-16f) * 0.25f;
#pragma unroll
  for (int p = 0; p < 2; ++p) {
    const int d = lane + p * 32;
    const float v = row[d] * c0 + row[64 + d] * c1 + row[128 + d] * c2 + row[192 + d] * c3;
    atomicAdd(&out[(size_t)dst * 64 + d], v);
  }
}

// ---------------------------------------------------------------------------
extern "C" void kernel_launch(void* const* d_in, const int* in_sizes, int n_in,
                              void* d_out, int out_size, void* d_ws, size_t ws_size,
                              hipStream_t stream) {
  const float* x_src = (const float*)d_in[0];
  const float* x_dst = (const float*)d_in[1];
  const int*   ei    = (const int*)d_in[2];
  const float* ea    = (const float*)d_in[3];
  const float* gamma = (const float*)d_in[4];
  const float* beta  = (const float*)d_in[5];
  const float* W_l   = (const float*)d_in[6];
  const float* b_l   = (const float*)d_in[7];
  const float* W_r   = (const float*)d_in[8];
  const float* b_r   = (const float*)d_in[9];
  const float* W_e   = (const float*)d_in[10];
  const float* att   = (const float*)d_in[11];
  const float* bias  = (const float*)d_in[12];
  float* out = (float*)d_out;

  // Workspace layout
  char* ws = (char*)d_ws;
  float*    hl    = (float*)ws;    ws += (size_t)NN * 256 * sizeof(float);
  float*    hr    = (float*)ws;    ws += (size_t)NN * 256 * sizeof(float);
  float*    alpha = (float*)ws;    ws += (size_t)EE * HEADS * sizeof(float);
  unsigned* amax  = (unsigned*)ws; ws += (size_t)NN * HEADS * sizeof(unsigned);
  float*    denom = (float*)ws;

  init_kernel<<<(NN * DDIM + 255) / 256, 256, 0, stream>>>(out, bias, amax, denom);
  ln_wmma_gemm<<<NN / 16, 256, 0, stream>>>(x_src, gamma, beta, W_l, b_l, hl);
  ln_wmma_gemm<<<NN / 16, 256, 0, stream>>>(x_dst, gamma, beta, W_r, b_r, hr);
  edge_score_kernel<<<EE / 8, 256, 0, stream>>>(hl, hr, ei, ea, W_e, att, alpha, amax);
  edge_exp_kernel<<<(EE * HEADS + 255) / 256, 256, 0, stream>>>(ei, alpha, amax, denom);
  edge_agg_kernel<<<EE / 8, 256, 0, stream>>>(hl, ei, alpha, denom, out);
}
